// InvariantPointAttentionV42_58497454571785
// MI455X (gfx1250) — matrix operand order repaired and checked by reading, not compiled, not run on hardware
//
#include <hip/hip_runtime.h>
#include <hip/hip_bf16.h>

typedef __attribute__((ext_vector_type(16))) __bf16 v16bf;
typedef __attribute__((ext_vector_type(8)))  __bf16 v8bf;
typedef __attribute__((ext_vector_type(8)))  float  v8f;

#define N_RES 768
#define C_DIM 384
#define H_N   12
#define CH    32
#define HP3   144
#define DP_N  128
#define KAUG  64    // 32 scalar + 12 point + 20 pad
#define VAUG  48    // 32 scalar + 12 point + 4 pad
#define KPAD  160   // HP3 padded up to a multiple of 32

// ---------------------------------------------------------------------------
// WMMA fragment loaders (CDNA5 wave32 layouts, CDNA5 ISA §7.12.2)
// A 16x32 bf16 per lane: two contiguous K-runs of 8 at (half*8) and (16+half*8).
// B 32x16 bf16 per lane: two K-runs of 8 at (half*8) and (16+half*8), column n = lane&15.
// C/D f32: element r -> M = r + 8*(lane>>4), N = lane&15.
// All loads unconditional + vectorized; K multiple of 32, rows 16B-aligned.
// ---------------------------------------------------------------------------

__device__ __forceinline__ v16bf a_frag_f32(const float* __restrict__ A, int lda,
                                            int row0, int k0) {
  int lane = threadIdx.x & 31;
  int m    = row0 + (lane & 15);
  int half = lane >> 4;
  const float* p = A + m * lda + k0 + half * 8;
  float4 x0 = ((const float4*)p)[0];
  float4 x1 = ((const float4*)p)[1];
  float4 x2 = ((const float4*)(p + 16))[0];
  float4 x3 = ((const float4*)(p + 16))[1];
  v16bf a;
  a[0]  = (__bf16)x0.x; a[1]  = (__bf16)x0.y; a[2]  = (__bf16)x0.z; a[3]  = (__bf16)x0.w;
  a[4]  = (__bf16)x1.x; a[5]  = (__bf16)x1.y; a[6]  = (__bf16)x1.z; a[7]  = (__bf16)x1.w;
  a[8]  = (__bf16)x2.x; a[9]  = (__bf16)x2.y; a[10] = (__bf16)x2.z; a[11] = (__bf16)x2.w;
  a[12] = (__bf16)x3.x; a[13] = (__bf16)x3.y; a[14] = (__bf16)x3.z; a[15] = (__bf16)x3.w;
  return a;
}

// B[k][n] = W[n][k] (weights row-major (Nout,K)): two runs of 8 fp32 -> bf16
__device__ __forceinline__ v16bf w_frag_f32(const float* __restrict__ W, int ldw,
                                            int col0, int k0) {
  int lane = threadIdx.x & 31;
  int n    = col0 + (lane & 15);
  int half = lane >> 4;
  const float* p = W + n * ldw + k0 + half * 8;
  v16bf b;
#pragma unroll
  for (int i = 0; i < 4; ++i) {
    float4 x = ((const float4*)(p + (i >> 1) * 16))[i & 1];
    b[4*i]   = (__bf16)x.x; b[4*i+1] = (__bf16)x.y;
    b[4*i+2] = (__bf16)x.z; b[4*i+3] = (__bf16)x.w;
  }
  return b;
}

// bf16 A (lda multiple of 8): two 16-byte runs
__device__ __forceinline__ v16bf a_frag_bf16(const __bf16* __restrict__ A, int lda,
                                             int row0, int k0) {
  int lane = threadIdx.x & 31;
  int m    = row0 + (lane & 15);
  int half = lane >> 4;
  const __bf16* p = A + m * lda + k0 + half * 8;
  v8bf lo = *(const v8bf*)p;
  v8bf hi = *(const v8bf*)(p + 16);
  v16bf a;
#pragma unroll
  for (int i = 0; i < 8; ++i) { a[i] = lo[i]; a[8 + i] = hi[i]; }
  return a;
}

// B[k][n] = M[n][k] (bf16 row-major along k): two 16-byte runs
__device__ __forceinline__ v16bf bT_frag_bf16(const __bf16* __restrict__ M, int ldm,
                                              int col0, int k0) {
  int lane = threadIdx.x & 31;
  int n    = col0 + (lane & 15);
  int half = lane >> 4;
  const __bf16* p = M + n * ldm + k0 + half * 8;
  v8bf lo = *(const v8bf*)p;
  v8bf hi = *(const v8bf*)(p + 16);
  return __builtin_shufflevector(lo, hi, 0, 1, 2, 3, 4, 5, 6, 7, 8, 9, 10, 11, 12, 13, 14, 15);
}

// A fragment from fp32 probabilities in LDS (ld = N_RES)
__device__ __forceinline__ v16bf a_frag_lds(const float* __restrict__ s, int k0) {
  int lane = threadIdx.x & 31;
  int m    = lane & 15;
  int half = lane >> 4;
  const float* p = s + m * N_RES + k0 + half * 8;
  float4 x0 = ((const float4*)p)[0];
  float4 x1 = ((const float4*)p)[1];
  float4 x2 = ((const float4*)(p + 16))[0];
  float4 x3 = ((const float4*)(p + 16))[1];
  v16bf a;
  a[0]  = (__bf16)x0.x; a[1]  = (__bf16)x0.y; a[2]  = (__bf16)x0.z; a[3]  = (__bf16)x0.w;
  a[4]  = (__bf16)x1.x; a[5]  = (__bf16)x1.y; a[6]  = (__bf16)x1.z; a[7]  = (__bf16)x1.w;
  a[8]  = (__bf16)x2.x; a[9]  = (__bf16)x2.y; a[10] = (__bf16)x2.z; a[11] = (__bf16)x2.w;
  a[12] = (__bf16)x3.x; a[13] = (__bf16)x3.y; a[14] = (__bf16)x3.z; a[15] = (__bf16)x3.w;
  return a;
}

// ---------------------------------------------------------------------------
// Generic GEMM: C[M,NOUT] = A[M,K] @ W[NOUT,K]^T + bias (+ add), bf16 WMMA.
// One wave per 16x(16*NTILES) strip; A fragment reused across NTILES B tiles.
// ---------------------------------------------------------------------------
template <int NOUT, int K, bool HAS_ADD, int NTILES>
__global__ void __launch_bounds__(32)
ipa_wmma_gemm(const float* __restrict__ A, const float* __restrict__ W,
              const float* __restrict__ bias, const float* __restrict__ add,
              float* __restrict__ C) {
  int row0 = blockIdx.x * 16;
  int col0 = blockIdx.y * (16 * NTILES);
  v8f acc[NTILES] = {};
  for (int k0 = 0; k0 < K; k0 += 32) {
    v16bf a = a_frag_f32(A, K, row0, k0);
#pragma unroll
    for (int t = 0; t < NTILES; ++t) {
      v16bf b = w_frag_f32(W, K, col0 + 16 * t, k0);
      acc[t] = __builtin_amdgcn_wmma_f32_16x16x32_bf16(false, a, false, b,
                                                       (short)0, acc[t],
                                                       false, false);
    }
  }
  int lane = threadIdx.x & 31;
  int half = lane >> 4;
#pragma unroll
  for (int t = 0; t < NTILES; ++t) {
    int n = col0 + 16 * t + (lane & 15);
    float bv = bias[n];
#pragma unroll
    for (int r = 0; r < 8; ++r) {
      int m = row0 + r + 8 * half;
      float vv = acc[t][r] + bv;
      if (HAS_ADD) vv += add[m * NOUT + n];
      C[m * NOUT + n] = vv;
    }
  }
}

// ---------------------------------------------------------------------------
// Rotation + translation applied in place; pts: [N][48][3]
// ---------------------------------------------------------------------------
__global__ void ipa_point_transform(float* __restrict__ pts,
                                    const float* __restrict__ rot,
                                    const float* __restrict__ trans) {
  int idx = blockIdx.x * blockDim.x + threadIdx.x;
  if (idx >= N_RES * 48) return;
  int n = idx / 48;
  float* p = pts + n * HP3 + (idx % 48) * 3;
  const float* R = rot + n * 9;
  float d0 = p[0], d1 = p[1], d2 = p[2];
  p[0] = d0 * R[0] + d1 * R[3] + d2 * R[6] + trans[n * 3 + 0];
  p[1] = d0 * R[1] + d1 * R[4] + d2 * R[7] + trans[n * 3 + 1];
  p[2] = d0 * R[2] + d1 * R[5] + d2 * R[8] + trans[n * 3 + 2];
}

__global__ void ipa_sqsum(const float* __restrict__ g, float* __restrict__ out) {
  int idx = blockIdx.x * blockDim.x + threadIdx.x;
  if (idx >= N_RES * H_N) return;
  int n = idx / H_N, h = idx % H_N;
  float s = 0.0f;
#pragma unroll
  for (int j = 0; j < 12; ++j) {
    float x = g[n * HP3 + h * 12 + j];
    s += x * x;
  }
  out[idx] = s;
}

// ---------------------------------------------------------------------------
// Pack augmented bf16 operands.
//   Qa[h][n][0:32]=scale*q, [32:44]=scale*q_g, rest 0   (row-major, ld=KAUG)
//   Ka analogous, unscaled.
//   VaT[h][c][n]  TRANSPOSED (ld=N_RES) so P·V B-fragments are contiguous.
// ---------------------------------------------------------------------------
__global__ void ipa_pack(const float* __restrict__ q, const float* __restrict__ k,
                         const float* __restrict__ v, const float* __restrict__ qg,
                         const float* __restrict__ kg, const float* __restrict__ vg,
                         __bf16* __restrict__ Qa, __bf16* __restrict__ Ka,
                         __bf16* __restrict__ VaT, float scale) {
  int idx = blockIdx.x * blockDim.x + threadIdx.x;
  if (idx >= H_N * N_RES) return;
  int h = idx / N_RES, n = idx % N_RES;
  __bf16* qo = Qa + (h * N_RES + n) * KAUG;
  __bf16* ko = Ka + (h * N_RES + n) * KAUG;
  __bf16* vt = VaT + h * VAUG * N_RES + n;
#pragma unroll 4
  for (int c = 0; c < 32; ++c) {
    qo[c] = (__bf16)(q[n * C_DIM + h * CH + c] * scale);
    ko[c] = (__bf16)(k[n * C_DIM + h * CH + c]);
    vt[c * N_RES] = (__bf16)(v[n * C_DIM + h * CH + c]);
  }
#pragma unroll
  for (int j = 0; j < 12; ++j) {
    qo[32 + j] = (__bf16)(qg[n * HP3 + h * 12 + j] * scale);
    ko[32 + j] = (__bf16)(kg[n * HP3 + h * 12 + j]);
    vt[(32 + j) * N_RES] = (__bf16)(vg[n * HP3 + h * 12 + j]);
  }
#pragma unroll
  for (int j = 44; j < KAUG; ++j) { qo[j] = (__bf16)0.0f; ko[j] = (__bf16)0.0f; }
#pragma unroll
  for (int j = 44; j < VAUG; ++j) vt[j * N_RES] = (__bf16)0.0f;
}

// Zero-pad Wpo (384x144) into (384x160)
__global__ void ipa_pad_wpo(const float* __restrict__ W, float* __restrict__ Wp) {
  int i = blockIdx.x * blockDim.x + threadIdx.x;
  if (i >= C_DIM * KPAD) return;
  int o = i / KPAD, kk = i % KPAD;
  Wp[i] = (kk < HP3) ? W[o * HP3 + kk] : 0.0f;
}

__global__ void ipa_zero(float* __restrict__ p, int n) {
  int i = blockIdx.x * blockDim.x + threadIdx.x;
  if (i < n) p[i] = 0.0f;
}

// ---------------------------------------------------------------------------
// Pair bias: out[h][n][m] = dot(pair[n][m][:], Wpb[h][:]) + bpb[h]
// Bandwidth-bound stream of 302 MB: one wave per (n,m), float4 per lane.
// ---------------------------------------------------------------------------
__global__ void __launch_bounds__(256)
ipa_pair_bias(const float* __restrict__ pair, const float* __restrict__ Wpb,
              const float* __restrict__ bpb, float* __restrict__ out) {
  __shared__ float sW[H_N * DP_N];
  __shared__ float sb[H_N];
  for (int i = threadIdx.x; i < H_N * DP_N; i += blockDim.x) sW[i] = Wpb[i];
  if (threadIdx.x < H_N) sb[threadIdx.x] = bpb[threadIdx.x];
  __syncthreads();

  int wave = threadIdx.x >> 5, lane = threadIdx.x & 31;
  long pairIdx = (long)blockIdx.x * 8 + wave;
  if (pairIdx >= (long)N_RES * N_RES) return;

  const float* p = pair + pairIdx * DP_N + lane * 4;
  __builtin_prefetch(p + 8 * DP_N, 0, 1);
  float4 x = *(const float4*)p;

  float acc[H_N];
#pragma unroll
  for (int h = 0; h < H_N; ++h) {
    const float* w = sW + h * DP_N + lane * 4;
    acc[h] = x.x * w[0] + x.y * w[1] + x.z * w[2] + x.w * w[3];
  }
#pragma unroll
  for (int off = 16; off > 0; off >>= 1)
#pragma unroll
    for (int h = 0; h < H_N; ++h) acc[h] += __shfl_down(acc[h], off, 32);

  if (lane == 0) {
    int n = (int)(pairIdx / N_RES), m = (int)(pairIdx % N_RES);
#pragma unroll
    for (int h = 0; h < H_N; ++h)
      out[((long)h * N_RES + n) * N_RES + m] = acc[h] + sb[h];
  }
}

// ---------------------------------------------------------------------------
// Attention core: one wave per (head, 16 query rows).
//   S = Qa·Ka^T (WMMA, K=64) + pair_bias - 0.5*scale*(q2+k2) -> 48KB LDS
//   vectorized in-place softmax; O = P·VaT (WMMA over 768).
// ---------------------------------------------------------------------------
__global__ void __launch_bounds__(32)
ipa_attention(const __bf16* __restrict__ Qa, const __bf16* __restrict__ Ka,
              const __bf16* __restrict__ VaT, const float* __restrict__ pb,
              const float* __restrict__ q2, const float* __restrict__ k2,
              float* __restrict__ wscalar, float* __restrict__ wpoints,
              float scale) {
  __shared__ float s_logits[16 * N_RES];  // 48 KB of the WGP's 320 KB

  int qt = blockIdx.x, h = blockIdx.y;
  int qrow0 = qt * 16;
  int lane = threadIdx.x & 31;
  int half = lane >> 4;

  const __bf16* Qh = Qa + h * N_RES * KAUG;
  const __bf16* Kh = Ka + h * N_RES * KAUG;
  const __bf16* Vh = VaT + h * VAUG * N_RES;
  const float*  pbh = pb + (size_t)h * N_RES * N_RES;

  v16bf aq0 = a_frag_bf16(Qh, KAUG, qrow0, 0);
  v16bf aq1 = a_frag_bf16(Qh, KAUG, qrow0, 32);

  float q2v[8];
#pragma unroll
  for (int r = 0; r < 8; ++r) q2v[r] = q2[(qrow0 + r + 8 * half) * H_N + h];

  // ---- logits ----
  for (int mt = 0; mt < N_RES / 16; ++mt) {
    v8f acc = {};
    v16bf b0 = bT_frag_bf16(Kh, KAUG, mt * 16, 0);
    v16bf b1 = bT_frag_bf16(Kh, KAUG, mt * 16, 32);
    acc = __builtin_amdgcn_wmma_f32_16x16x32_bf16(false, aq0, false, b0,
                                                  (short)0, acc, false, false);
    acc = __builtin_amdgcn_wmma_f32_16x16x32_bf16(false, aq1, false, b1,
                                                  (short)0, acc, false, false);
    int m = mt * 16 + (lane & 15);
    float k2m = k2[m * H_N + h];
#pragma unroll
    for (int r = 0; r < 8; ++r) {
      int qrow = qrow0 + r + 8 * half;
      float logit = acc[r] + pbh[qrow * N_RES + m]
                  - 0.5f * scale * (q2v[r] + k2m);
      s_logits[(r + 8 * half) * N_RES + m] = logit;
    }
  }
  __syncthreads();

  // ---- softmax: lanes (l, l^16) share row l&15, each owns 384 cols.
  //      All passes vectorized float4 (ds_load_b128 / ds_store_b128). ----
  {
    float4* row = (float4*)(s_logits + (lane & 15) * N_RES + half * (N_RES / 2));
    float4 m4 = row[0];
    for (int c = 1; c < N_RES / 8; ++c) {
      float4 x = row[c];
      m4.x = fmaxf(m4.x, x.x); m4.y = fmaxf(m4.y, x.y);
      m4.z = fmaxf(m4.z, x.z); m4.w = fmaxf(m4.w, x.w);
    }
    float mx = fmaxf(fmaxf(m4.x, m4.y), fmaxf(m4.z, m4.w));
    mx = fmaxf(mx, __shfl_xor(mx, 16, 32));

    float4 s4 = make_float4(0.f, 0.f, 0.f, 0.f);
    for (int c = 0; c < N_RES / 8; ++c) {
      float4 x = row[c];
      x.x = __expf(x.x - mx); x.y = __expf(x.y - mx);
      x.z = __expf(x.z - mx); x.w = __expf(x.w - mx);
      row[c] = x;
      s4.x += x.x; s4.y += x.y; s4.z += x.z; s4.w += x.w;
    }
    float sum = (s4.x + s4.y) + (s4.z + s4.w);
    sum += __shfl_xor(sum, 16, 32);
    float inv = 1.0f / sum;
    for (int c = 0; c < N_RES / 8; ++c) {
      float4 x = row[c];
      x.x *= inv; x.y *= inv; x.z *= inv; x.w *= inv;
      row[c] = x;
    }
  }
  __syncthreads();

  // ---- O = P·V : 3 column tiles (48 cols), K over 768 ----
  for (int nt = 0; nt < 3; ++nt) {
    v8f acc = {};
    for (int k0 = 0; k0 < N_RES; k0 += 32) {
      v16bf a = a_frag_lds(s_logits, k0);
      v16bf b = bT_frag_bf16(Vh, N_RES, nt * 16, k0);  // VaT[col][k] contiguous
      acc = __builtin_amdgcn_wmma_f32_16x16x32_bf16(false, a, false, b,
                                                    (short)0, acc, false, false);
    }
    int col = nt * 16 + (lane & 15);
#pragma unroll
    for (int r = 0; r < 8; ++r) {
      int qrow = qrow0 + r + 8 * half;
      if (col < 32) {
        wscalar[qrow * C_DIM + h * CH + col] = acc[r];
      } else if (col < 44) {
        wpoints[qrow * KPAD + h * 12 + (col - 32)] = acc[r];
      }
    }
  }
}

// ---------------------------------------------------------------------------
// Residual + LayerNorm, one wave per row.
// ---------------------------------------------------------------------------
__global__ void __launch_bounds__(32)
ipa_layernorm(const float* __restrict__ x, const float* __restrict__ lw,
              const float* __restrict__ lb, float* __restrict__ out) {
  int n = blockIdx.x, lane = threadIdx.x;
  float s = 0.0f;
  for (int c = lane; c < C_DIM; c += 32) s += x[n * C_DIM + c];
#pragma unroll
  for (int off = 16; off > 0; off >>= 1) s += __shfl_xor(s, off, 32);
  float mu = s * (1.0f / C_DIM);
  float vs = 0.0f;
  for (int c = lane; c < C_DIM; c += 32) {
    float d = x[n * C_DIM + c] - mu;
    vs += d * d;
  }
#pragma unroll
  for (int off = 16; off > 0; off >>= 1) vs += __shfl_xor(vs, off, 32);
  float inv = rsqrtf(vs * (1.0f / C_DIM) + 1e-5f);
  for (int c = lane; c < C_DIM; c += 32)
    out[n * C_DIM + c] = (x[n * C_DIM + c] - mu) * inv * lw[c] + lb[c];
}

// ---------------------------------------------------------------------------
extern "C" void kernel_launch(void* const* d_in, const int* in_sizes, int n_in,
                              void* d_out, int out_size, void* d_ws, size_t ws_size,
                              hipStream_t stream) {
  (void)in_sizes; (void)n_in; (void)out_size; (void)ws_size;
  const float* single = (const float*)d_in[0];
  const float* pair   = (const float*)d_in[1];
  const float* rot    = (const float*)d_in[2];
  const float* trans  = (const float*)d_in[3];
  const float* Wq  = (const float*)d_in[4];   const float* bq  = (const float*)d_in[5];
  const float* Wk  = (const float*)d_in[6];   const float* bk  = (const float*)d_in[7];
  const float* Wv  = (const float*)d_in[8];   const float* bv  = (const float*)d_in[9];
  const float* Wpb = (const float*)d_in[10];  const float* bpb = (const float*)d_in[11];
  const float* Wqp = (const float*)d_in[12];  const float* bqp = (const float*)d_in[13];
  const float* Wkp = (const float*)d_in[14];  const float* bkp = (const float*)d_in[15];
  const float* Wvp = (const float*)d_in[16];  const float* bvp = (const float*)d_in[17];
  const float* Wo  = (const float*)d_in[18];  const float* bo  = (const float*)d_in[19];
  const float* Wpo = (const float*)d_in[20];  const float* bpo = (const float*)d_in[21];
  const float* lnw = (const float*)d_in[22];  const float* lnb = (const float*)d_in[23];
  float* out = (float*)d_out;

  size_t off = 0;
  auto alloc = [&](size_t bytes) -> void* {
    void* p = (char*)d_ws + off;
    off += (bytes + 255) & ~(size_t)255;
    return p;
  };
  float* q    = (float*)alloc((size_t)N_RES * C_DIM * 4);
  float* k    = (float*)alloc((size_t)N_RES * C_DIM * 4);
  float* v    = (float*)alloc((size_t)N_RES * C_DIM * 4);
  float* qg   = (float*)alloc((size_t)N_RES * HP3 * 4);
  float* kg   = (float*)alloc((size_t)N_RES * HP3 * 4);
  float* vg   = (float*)alloc((size_t)N_RES * HP3 * 4);
  float* q2   = (float*)alloc((size_t)N_RES * H_N * 4);
  float* k2   = (float*)alloc((size_t)N_RES * H_N * 4);
  __bf16* Qa  = (__bf16*)alloc((size_t)H_N * N_RES * KAUG * 2);
  __bf16* Ka  = (__bf16*)alloc((size_t)H_N * N_RES * KAUG * 2);
  __bf16* VaT = (__bf16*)alloc((size_t)H_N * VAUG * N_RES * 2);
  float* pb   = (float*)alloc((size_t)H_N * N_RES * N_RES * 4);
  float* wsc  = (float*)alloc((size_t)N_RES * C_DIM * 4);
  float* wpt  = (float*)alloc((size_t)N_RES * KPAD * 4);
  float* wpoP = (float*)alloc((size_t)C_DIM * KPAD * 4);
  float* tmp  = (float*)alloc((size_t)N_RES * C_DIM * 4);
  float* xr   = (float*)alloc((size_t)N_RES * C_DIM * 4);

  const float scale = 0.17677669529663687f;  // 32^-0.5

  // 1) scalar + point projections (bf16 WMMA GEMMs, 16x32 strips)
  dim3 gC(N_RES / 16, C_DIM / 32), gP(N_RES / 16, HP3 / 16);
  ipa_wmma_gemm<C_DIM, C_DIM, false, 2><<<gC, 32, 0, stream>>>(single, Wq,  bq,  nullptr, q);
  ipa_wmma_gemm<C_DIM, C_DIM, false, 2><<<gC, 32, 0, stream>>>(single, Wk,  bk,  nullptr, k);
  ipa_wmma_gemm<C_DIM, C_DIM, false, 2><<<gC, 32, 0, stream>>>(single, Wv,  bv,  nullptr, v);
  ipa_wmma_gemm<HP3,   C_DIM, false, 1><<<gP, 32, 0, stream>>>(single, Wqp, bqp, nullptr, qg);
  ipa_wmma_gemm<HP3,   C_DIM, false, 1><<<gP, 32, 0, stream>>>(single, Wkp, bkp, nullptr, kg);
  ipa_wmma_gemm<HP3,   C_DIM, false, 1><<<gP, 32, 0, stream>>>(single, Wvp, bvp, nullptr, vg);

  // 2) rigid transform + squared norms
  int tN = N_RES * 48;
  ipa_point_transform<<<(tN + 255) / 256, 256, 0, stream>>>(qg, rot, trans);
  ipa_point_transform<<<(tN + 255) / 256, 256, 0, stream>>>(kg, rot, trans);
  ipa_point_transform<<<(tN + 255) / 256, 256, 0, stream>>>(vg, rot, trans);
  int sN = N_RES * H_N;
  ipa_sqsum<<<(sN + 255) / 256, 256, 0, stream>>>(qg, q2);
  ipa_sqsum<<<(sN + 255) / 256, 256, 0, stream>>>(kg, k2);

  // 3) pack augmented bf16 operands; zero-pad wpt; pad Wpo to K=160
  ipa_pack<<<(H_N * N_RES + 255) / 256, 256, 0, stream>>>(q, k, v, qg, kg, vg,
                                                          Qa, Ka, VaT, scale);
  ipa_zero<<<(N_RES * KPAD + 255) / 256, 256, 0, stream>>>(wpt, N_RES * KPAD);
  ipa_pad_wpo<<<(C_DIM * KPAD + 255) / 256, 256, 0, stream>>>(Wpo, wpoP);

  // 4) pair bias (bandwidth-bound: 302 MB @ 23.3 TB/s ~ 13 us floor)
  ipa_pair_bias<<<(N_RES * N_RES) / 8, 256, 0, stream>>>(pair, Wpb, bpb, pb);

  // 5) fused attention (WMMA logits + softmax + WMMA P·V)
  dim3 gA(N_RES / 16, H_N);
  ipa_attention<<<gA, 32, 0, stream>>>(Qa, Ka, VaT, pb, q2, k2, wsc, wpt, scale);

  // 6) point projection (+ weighted_scalar add), then output proj (+ residual)
  ipa_wmma_gemm<C_DIM, KPAD,  true, 2><<<gC, 32, 0, stream>>>(wpt, wpoP, bpo, wsc,    tmp);
  ipa_wmma_gemm<C_DIM, C_DIM, true, 2><<<gC, 32, 0, stream>>>(tmp, Wo,   bo,  single, xr);

  // 7) LayerNorm
  ipa_layernorm<<<N_RES, 32, 0, stream>>>(xr, lnw, lnb, out);
}
